// MultiHeadedAttention_37769942401072
// MI455X (gfx1250) — matrix operand order repaired and checked by reading, not compiled, not run on hardware
//
#include <hip/hip_runtime.h>

// ---------------------------------------------------------------------------
// MultiHeadedAttention on MI455X (gfx1250, wave32, WMMA bf16 16x16x32)
// B=2, S=2048, D=1024, H=16, DH=64.
// bf16 WMMA everywhere (f32 accumulate), flash online softmax, LDS-staged
// K/V tiles shared by the 4 waves of a workgroup (double buffered).
// ---------------------------------------------------------------------------

typedef __attribute__((ext_vector_type(16))) __bf16 v16bf;
typedef __attribute__((ext_vector_type(8)))  __bf16 v8bf;
typedef __attribute__((ext_vector_type(8)))  float  v8f;

#define DEV __device__ __forceinline__

constexpr int B_ = 2, S_ = 2048, D_ = 1024, H_ = 16, DH_ = 64;

union FragU { v16bf v; v8bf h[2]; };

// A-fragment (16x32 bf16, M x K), source row-major [M][K] with stride ld.
// Layout: K = 16*hi + j (lanes 0-15 hold K=0..15, lanes 16-31 hold K=16..31).
DEV v16bf load_frag_A(const __bf16* base, int ld) {
  const int lane = threadIdx.x & 31;
  const __bf16* p = base + (lane & 15) * ld + (lane >> 4) * 16;
  FragU f;
  f.h[0] = *(const v8bf*)p;         // K = hi*16 + 0..7
  f.h[1] = *(const v8bf*)(p + 8);   // K = hi*16 + 8..15
  return f.v;
}

// A-fragment from f32 source (convert to bf16 in-register).
DEV v16bf load_frag_A_f32(const float* base, int ld) {
  const int lane = threadIdx.x & 31;
  const float* p = base + (lane & 15) * ld + (lane >> 4) * 16;
  v16bf f;
#pragma unroll
  for (int j = 0; j < 8; ++j) f[j] = (__bf16)p[j];
#pragma unroll
  for (int j = 0; j < 8; ++j) f[8 + j] = (__bf16)p[8 + j];
  return f;
}

// B-fragment (32x16 bf16, K x N) where B(k,n) = M[n][k], M row-major stride ld.
// Layout (same contiguous k order as load_frag_A): lanes 0-15 hold K=0..15, lanes 16-31 hold K=16..31, N=lane%16.
DEV v16bf load_frag_B(const __bf16* base, int ld) {
  const int lane = threadIdx.x & 31;
  const __bf16* p = base + (lane & 15) * ld + (lane >> 4) * 16;
  FragU f;
  f.h[0] = *(const v8bf*)p;
  f.h[1] = *(const v8bf*)(p + 8);
  return f.v;
}

DEV v8f wmma_bf16(v16bf a, v16bf b, v8f c) {
  // (neg_a, A, neg_b, B, c_mod, C, reuse_a, reuse_b)
  return __builtin_amdgcn_wmma_f32_16x16x32_bf16(false, a, false, b,
                                                 (short)0, c, false, false);
}

// ---------------------------------------------------------------------------
// Kernel 1: convert the four DxD weight matrices f32 -> bf16 (order q,k,v,o).
// ---------------------------------------------------------------------------
__global__ __launch_bounds__(256) void wconv_kernel(const float* __restrict__ Wq,
                                                    const float* __restrict__ Wk,
                                                    const float* __restrict__ Wv,
                                                    const float* __restrict__ Wo,
                                                    __bf16* __restrict__ out) {
  const int i = blockIdx.x * 256 + threadIdx.x;      // 0 .. 4*D*D-1
  const int which = i >> 20;                         // D*D == 1<<20
  const int off   = i & ((1 << 20) - 1);
  const float* src = (which == 0) ? Wq : (which == 1) ? Wk : (which == 2) ? Wv : Wo;
  out[i] = (__bf16)src[off];
}

// ---------------------------------------------------------------------------
// Kernel 2: fused QKV projection.  Y[m,n] = sum_d X[m,d]*W[n,d] + b[n].
// 4 waves/block; wave -> one 16(M)x64(N) strip.  blockIdx.z picks Q/K/V.
//  Q -> Qb[b,h,s,dh] * (1/sqrt(DH)),  K -> Kb[b,h,s,dh],  V -> Vt[b,h,dh,s].
// ---------------------------------------------------------------------------
__global__ __launch_bounds__(128) void proj_kernel(
    const float* __restrict__ Xq, const float* __restrict__ Xk,
    const float* __restrict__ Xv,
    const __bf16* __restrict__ Wqb, const __bf16* __restrict__ Wkb,
    const __bf16* __restrict__ Wvb,
    const float* __restrict__ bq, const float* __restrict__ bk,
    const float* __restrict__ bv,
    __bf16* __restrict__ Qb, __bf16* __restrict__ Kb, __bf16* __restrict__ Vt) {
  const int which = blockIdx.z;                 // 0=q 1=k 2=v
  const int wave  = threadIdx.x >> 5;
  const int lane  = threadIdx.x & 31;
  const int mtile = blockIdx.x * 4 + wave;      // 0..255 (rows of 16 tokens)
  const int nbase = blockIdx.y * 64;            // output-feature strip
  const float*  X = (which == 0) ? Xq : (which == 1) ? Xk : Xv;
  const __bf16* W = (which == 0) ? Wqb : (which == 1) ? Wkb : Wvb;
  const float* bias = (which == 0) ? bq : (which == 1) ? bk : bv;

  const float* Xrow = X + (size_t)mtile * 16 * D_ + (lane & 15) * D_;

  v8f acc[4] = {};
  for (int k0 = 0; k0 < D_; k0 += 32) {
    if (k0 + 32 < D_)  // gfx1250 global_prefetch of next A tile
      __builtin_prefetch(Xrow + k0 + 32, 0, 1);
    v16bf a = load_frag_A_f32(X + (size_t)mtile * 16 * D_ + k0, D_);
    v16bf bfr[4];
#pragma unroll
    for (int t = 0; t < 4; ++t)
      bfr[t] = load_frag_B(W + (size_t)(nbase + t * 16) * D_ + k0, D_);
#pragma unroll
    for (int t = 0; t < 4; ++t) acc[t] = wmma_bf16(a, bfr[t], acc[t]);
  }
  const int nlo = lane & 15, mhi = lane >> 4;
#pragma unroll
  for (int t = 0; t < 4; ++t) {
    const int n = nbase + t * 16 + nlo;
    const float bb = bias[n];
    const int h = n >> 6, dh = n & 63;
#pragma unroll
    for (int i = 0; i < 8; ++i) {
      const int m = mtile * 16 + i + mhi * 8;
      const int bb_idx = m >> 11, s = m & (S_ - 1);
      const float y = acc[t][i] + bb;
      if (which == 0)
        Qb[(((size_t)(bb_idx * H_ + h) * S_ + s) << 6) + dh] = (__bf16)(y * 0.125f);
      else if (which == 1)
        Kb[(((size_t)(bb_idx * H_ + h) * S_ + s) << 6) + dh] = (__bf16)y;
      else
        Vt[((size_t)(bb_idx * H_ + h) * DH_ + dh) * S_ + s] = (__bf16)y;
    }
  }
}

// ---------------------------------------------------------------------------
// Kernel 3: flash attention.  4 waves/block share one (b,h); each wave owns a
// 16-query tile.  K (32x64) and V^T (64x32) tiles are staged in LDS (double
// buffered: global->VGPR issued before compute, ds_store after a barrier).
// Scores computed transposed St[k,q] = K x Q^T (softmax = 8 in-lane ops + one
// half-wave shuffle); context transposed Ct[d,q] = V^T x P.
// ---------------------------------------------------------------------------
DEV void stage_load(const __bf16* Kh, const __bf16* Vh, int kb, int t,
                    v8bf& rk0, v8bf& rk1, v8bf& rv0, v8bf& rv1) {
  const __bf16* kp = Kh + (size_t)(kb + (t >> 2)) * DH_ + (t & 3) * 16;
  rk0 = *(const v8bf*)kp;
  rk1 = *(const v8bf*)(kp + 8);
  const __bf16* vp = Vh + (size_t)(t >> 1) * S_ + kb + (t & 1) * 16;
  rv0 = *(const v8bf*)vp;
  rv1 = *(const v8bf*)(vp + 8);
}

DEV void stage_store(__bf16* lK, __bf16* lV, int t,
                     v8bf rk0, v8bf rk1, v8bf rv0, v8bf rv1) {
  __bf16* kd = lK + (t >> 2) * DH_ + (t & 3) * 16;
  *(v8bf*)kd = rk0;
  *(v8bf*)(kd + 8) = rk1;
  __bf16* vd = lV + (t >> 1) * 32 + (t & 1) * 16;
  *(v8bf*)vd = rv0;
  *(v8bf*)(vd + 8) = rv1;
}

__global__ __launch_bounds__(128) void attn_kernel(
    const __bf16* __restrict__ Qb, const __bf16* __restrict__ Kb,
    const __bf16* __restrict__ Vt, const unsigned char* __restrict__ mask,
    __bf16* __restrict__ ctx, float* __restrict__ ml) {
  __shared__ __bf16 lK[2][32 * DH_];   // 2 x 4 KB
  __shared__ __bf16 lV[2][DH_ * 32];   // 2 x 4 KB

  const int tid = threadIdx.x;
  const int wave = tid >> 5;
  const int qt = blockIdx.x * 4 + wave;  // 0..127
  const int h = blockIdx.y, b = blockIdx.z;
  const int lane = tid & 31;
  const int qlane = lane & 15;           // N index = q within tile
  const int khalf = lane >> 4;

  const __bf16* Qh = Qb + ((size_t)(b * H_ + h) * S_ + qt * 16) * DH_;
  const __bf16* Kh = Kb + (size_t)(b * H_ + h) * S_ * DH_;
  const __bf16* Vh = Vt + (size_t)(b * H_ + h) * DH_ * S_;
  const unsigned char* mq =
      mask + (size_t)b * S_ * S_ + (size_t)(qt * 16 + qlane) * S_;

  // Q as B-fragments (B(d,q) = Q[q][d]); d = 0..31 and 32..63.
  const v16bf qf0 = load_frag_B(Qh, DH_);
  const v16bf qf1 = load_frag_B(Qh + 32, DH_);

  float m_run = -3.0e38f, l_run = 0.0f;
  v8f co[4] = {};  // Ct accumulators, d-tiles 0..3

  // prologue: stage first K/V tile into buffer 0
  v8bf rk0, rk1, rv0, rv1;
  stage_load(Kh, Vh, 0, tid, rk0, rk1, rv0, rv1);
  stage_store(&lK[0][0], &lV[0][0], tid, rk0, rk1, rv0, rv1);
  __syncthreads();

  for (int kb = 0; kb < S_; kb += 32) {
    const int cur = (kb >> 5) & 1;
    const bool more = (kb + 32) < S_;
    if (more) stage_load(Kh, Vh, kb + 32, tid, rk0, rk1, rv0, rv1);

    const __bf16* cK = &lK[cur][0];
    const __bf16* cV = &lV[cur][0];

    // two transposed 16x16 score tiles (k = kb..kb+15 and kb+16..kb+31)
    v16bf ka0 = load_frag_A(cK, DH_);
    v16bf ka1 = load_frag_A(cK + 32, DH_);
    v16bf kb0 = load_frag_A(cK + 16 * DH_, DH_);
    v16bf kb1 = load_frag_A(cK + 16 * DH_ + 32, DH_);
    v8f s0 = {}, s1 = {};
    s0 = wmma_bf16(ka0, qf0, s0);
    s0 = wmma_bf16(ka1, qf1, s0);
    s1 = wmma_bf16(kb0, qf0, s1);
    s1 = wmma_bf16(kb1, qf1, s1);

    // mask: St element (k,q): q = qlane, k = kb + i + 8*khalf (+16 tile 1).
    // Per lane the 8 bytes are contiguous -> two b64 loads.
    const unsigned long long m0 =
        *(const unsigned long long*)(mq + kb + 8 * khalf);
    const unsigned long long m1 =
        *(const unsigned long long*)(mq + kb + 16 + 8 * khalf);
#pragma unroll
    for (int i = 0; i < 8; ++i) {
      if ((m0 >> (8 * i)) & 0xffULL) s0[i] = -1e18f;
      if ((m1 >> (8 * i)) & 0xffULL) s1[i] = -1e18f;
    }

    // online softmax (per lane = per q column; combine the two lane halves)
    float mloc = -3.0e38f;
#pragma unroll
    for (int i = 0; i < 8; ++i) mloc = fmaxf(mloc, fmaxf(s0[i], s1[i]));
    mloc = fmaxf(mloc, __shfl_xor(mloc, 16, 32));
    const float m_new = fmaxf(m_run, mloc);
    const float scale = __expf(m_run - m_new);
    float pe[8], po[8], psum = 0.0f;
#pragma unroll
    for (int i = 0; i < 8; ++i) {
      pe[i] = __expf(s0[i] - m_new);
      po[i] = __expf(s1[i] - m_new);
      psum += pe[i] + po[i];
    }
    psum += __shfl_xor(psum, 16, 32);
    l_run = l_run * scale + psum;
    m_run = m_new;
#pragma unroll
    for (int t = 0; t < 4; ++t)
#pragma unroll
      for (int i = 0; i < 8; ++i) co[t][i] *= scale;

    // Relayout P (two C-layout tiles) into one 32x16 B-fragment via one
    // half-wave shuffle per VGPR, converting to bf16.
    v16bf pf;
#pragma unroll
    for (int j = 0; j < 8; ++j) {
      const float ax = __shfl_xor(pe[j], 16, 32);
      const float bx = __shfl_xor(po[j], 16, 32);
      pf[j]     = (khalf == 0) ? (__bf16)pe[j] : (__bf16)bx;
      pf[j + 8] = (khalf == 0) ? (__bf16)ax    : (__bf16)po[j];
    }
    // context: Ct[d,q] += A(V^T tile d) x P   (A-fragments from LDS, ld=32)
#pragma unroll
    for (int t = 0; t < 4; ++t) {
      v16bf va = load_frag_A(cV + t * 16 * 32, 32);
      co[t] = wmma_bf16(va, pf, co[t]);
    }

    if (more) {  // publish next tile into the other buffer
      __syncthreads();
      stage_store(&lK[cur ^ 1][0], &lV[cur ^ 1][0], tid, rk0, rk1, rv0, rv1);
      __syncthreads();
    }
  }

  const float inv_l = 1.0f / l_run;
  const int q = qt * 16 + qlane;
  __bf16* crow = ctx + (size_t)(b * S_ + q) * D_ + h * DH_;
#pragma unroll
  for (int t = 0; t < 4; ++t)
#pragma unroll
    for (int i = 0; i < 8; ++i)
      crow[t * 16 + i + 8 * khalf] = (__bf16)(co[t][i] * inv_l);

  if (lane < 16) {  // per-row (m,l) for head-0 recompute pass
    float* r = ml + (size_t)((b * H_ + h) * S_ + q) * 2;
    r[0] = m_run;
    r[1] = l_run;
  }
}

// ---------------------------------------------------------------------------
// Kernel 4: head-0 attention recompute -> top_attn (f32).  Normal orientation
// (M=q, N=k) so consecutive lanes store consecutive k.
// ---------------------------------------------------------------------------
__global__ __launch_bounds__(128) void attn0_kernel(
    const __bf16* __restrict__ Qb, const __bf16* __restrict__ Kb,
    const unsigned char* __restrict__ mask, const float* __restrict__ ml,
    float* __restrict__ top) {
  const int wave = threadIdx.x >> 5;
  const int qt = blockIdx.x;                 // 0..127
  const int ks = blockIdx.y * 4 + wave;      // 0..31 (64-wide k strips)
  const int b = blockIdx.z;
  const int lane = threadIdx.x & 31;
  const int nlo = lane & 15, mhi = lane >> 4;

  const __bf16* Qh = Qb + ((size_t)(b * H_) * S_ + qt * 16) * DH_;  // h = 0
  const __bf16* Kh = Kb + (size_t)(b * H_) * S_ * DH_;

  const v16bf a0 = load_frag_A(Qh, DH_);
  const v16bf a1 = load_frag_A(Qh + 32, DH_);
  v16bf bf[8];
#pragma unroll
  for (int t = 0; t < 4; ++t) {
    const int kb = ks * 64 + t * 16;
    bf[2 * t]     = load_frag_B(Kh + (size_t)kb * DH_, DH_);
    bf[2 * t + 1] = load_frag_B(Kh + (size_t)kb * DH_ + 32, DH_);
  }
  v8f s[4] = {};
#pragma unroll
  for (int t = 0; t < 4; ++t) {
    s[t] = wmma_bf16(a0, bf[2 * t], s[t]);
    s[t] = wmma_bf16(a1, bf[2 * t + 1], s[t]);
  }
#pragma unroll
  for (int t = 0; t < 4; ++t) {
    const int k = ks * 64 + t * 16 + nlo;
#pragma unroll
    for (int i = 0; i < 8; ++i) {
      const int q = qt * 16 + i + mhi * 8;
      const float* r = ml + (size_t)((b * H_) * S_ + q) * 2;
      float sv = s[t][i];
      if (mask[(size_t)b * S_ * S_ + (size_t)q * S_ + k]) sv = -1e18f;
      top[(size_t)b * S_ * S_ + (size_t)q * S_ + k] = __expf(sv - r[0]) / r[1];
    }
  }
}

// ---------------------------------------------------------------------------
// Kernel 5: output projection  out[m,n] = sum_d ctx[m,d]*Wo[n,d] + bo[n] (f32)
// ---------------------------------------------------------------------------
__global__ __launch_bounds__(128) void outproj_kernel(
    const __bf16* __restrict__ ctx, const __bf16* __restrict__ Wob,
    const float* __restrict__ bo, float* __restrict__ out) {
  const int wave = threadIdx.x >> 5;
  const int lane = threadIdx.x & 31;
  const int mtile = blockIdx.x * 4 + wave;
  const int nbase = blockIdx.y * 64;

  const __bf16* Arow = ctx + (size_t)mtile * 16 * D_ + (lane & 15) * D_;

  v8f acc[4] = {};
  for (int k0 = 0; k0 < D_; k0 += 32) {
    if (k0 + 32 < D_)
      __builtin_prefetch(Arow + k0 + 32, 0, 1);
    v16bf a = load_frag_A(ctx + (size_t)mtile * 16 * D_ + k0, D_);
    v16bf bfr[4];
#pragma unroll
    for (int t = 0; t < 4; ++t)
      bfr[t] = load_frag_B(Wob + (size_t)(nbase + t * 16) * D_ + k0, D_);
#pragma unroll
    for (int t = 0; t < 4; ++t) acc[t] = wmma_bf16(a, bfr[t], acc[t]);
  }
  const int nlo = lane & 15, mhi = lane >> 4;
#pragma unroll
  for (int t = 0; t < 4; ++t) {
    const int n = nbase + t * 16 + nlo;
    const float bb = bo[n];
#pragma unroll
    for (int i = 0; i < 8; ++i) {
      const int m = mtile * 16 + i + mhi * 8;
      out[(size_t)m * D_ + n] = acc[t][i] + bb;
    }
  }
}

// ---------------------------------------------------------------------------
extern "C" void kernel_launch(void* const* d_in, const int* in_sizes, int n_in,
                              void* d_out, int out_size, void* d_ws,
                              size_t ws_size, hipStream_t stream) {
  (void)in_sizes; (void)n_in; (void)out_size; (void)ws_size;
  const float* key   = (const float*)d_in[0];
  const float* value = (const float*)d_in[1];
  const float* query = (const float*)d_in[2];
  const unsigned char* mask = (const unsigned char*)d_in[3];  // jnp.bool_ = 1B
  const float* Wk = (const float*)d_in[4];
  const float* bk = (const float*)d_in[5];
  const float* Wv = (const float*)d_in[6];
  const float* bv = (const float*)d_in[7];
  const float* Wq = (const float*)d_in[8];
  const float* bq = (const float*)d_in[9];
  const float* Wo = (const float*)d_in[10];
  const float* bo = (const float*)d_in[11];

  const size_t MB = 1024 * 1024;
  const size_t DD = (size_t)D_ * D_;
  char* ws = (char*)d_ws;
  __bf16* Wb  = (__bf16*)(ws);            // 4*DD bf16 = 8 MB (order q,k,v,o)
  __bf16* Wqb = Wb;
  __bf16* Wkb = Wb + DD;
  __bf16* Wvb = Wb + 2 * DD;
  __bf16* Wob = Wb + 3 * DD;
  __bf16* Qb  = (__bf16*)(ws + 8 * MB);   // [B,H,S,DH] bf16, pre-scaled
  __bf16* Kb  = (__bf16*)(ws + 16 * MB);  // [B,H,S,DH] bf16
  __bf16* Vt  = (__bf16*)(ws + 24 * MB);  // [B,H,DH,S] bf16 (transposed)
  __bf16* ctx = (__bf16*)(ws + 32 * MB);  // [B,S,D]    bf16
  float*  ml  = (float*)(ws + 40 * MB);   // [B,H,S,2]  f32 (row max, row sum)

  float* out = (float*)d_out;                    // [B,S,D]
  float* top = out + (size_t)B_ * S_ * D_;       // [B,S,S]

  wconv_kernel<<<(unsigned)(4 * DD / 256), 256, 0, stream>>>(Wq, Wk, Wv, Wo, Wb);
  proj_kernel<<<dim3(64, 16, 3), 128, 0, stream>>>(query, key, value, Wqb, Wkb,
                                                   Wvb, bq, bk, bv, Qb, Kb, Vt);
  attn_kernel<<<dim3(32, H_, B_), 128, 0, stream>>>(Qb, Kb, Vt, mask, ctx, ml);
  attn0_kernel<<<dim3(128, 8, B_), 128, 0, stream>>>(Qb, Kb, mask, ml, top);
  outproj_kernel<<<dim3(64, 16), 128, 0, stream>>>(ctx, Wob, bo, out);
}
